// PerformerBlock_53128745451823
// MI455X (gfx1250) — matrix operand order repaired and checked by reading, not compiled, not run on hardware
//
#include <hip/hip_runtime.h>
#include <hip/hip_bf16.h>

typedef __attribute__((ext_vector_type(16))) _Float16 v16h;
typedef __attribute__((ext_vector_type(8)))  float    v8f;

#define BDIM   4
#define NSEQ   4096
#define CDIM   1024
#define HEADS  16
#define DHEAD  64
#define BN     (BDIM*NSEQ)      // 16384
#define MFEAT  266
#define MPAD   320              // padded feature count (multiple of 64)
#define EPS_K  1e-3f
#define LN_EPS 1e-5f
#define LDS_STRIDE 72           // 64 + 8 halfs pad, keeps 16B alignment

enum {
  EPI_PLAIN = 0,   // f16 store, row-major
  EPI_VSPLIT,      // v -> vT[b,h,d,n] f16, masked
  EPI_FEAT,        // relu+eps (zero pad cols) -> qf[b,h,n,m] f16
  EPI_FEATT,       // relu+eps (zero pad cols) -> kfT[b,h,m,n] f16
  EPI_CTXT,        // -> ctxT[b,h,d,m] f16
  EPI_OUTSCALE,    // *dinv, scatter to o[bn, h*64+d] f16
  EPI_BIASRES,     // + bias + residual -> f32
  EPI_GELU         // gelu(x+bias) -> f16
};

struct GP {
  const _Float16* A; const _Float16* Bm; void* D;
  int M, N, K, lda, ldb, ldd;
  long long sAhi, sAlo; int zdiv;      // A batch offset = (z/zdiv)*sAhi + (z%zdiv)*sAlo
  long long sBz, sDz;
  const float* bias; const float* res; const float* dinv;
  const unsigned char* mask;
};

// Async global -> LDS copy, 16B per lane (tracked by ASYNCcnt).
__device__ __forceinline__ void async_load_b128(unsigned ldsoff, const void* gptr) {
  asm volatile("global_load_async_to_lds_b128 %0, %1, off"
               :: "v"(ldsoff), "v"((unsigned long long)gptr)
               : "memory");
}
__device__ __forceinline__ void wait_async0() {
  asm volatile("s_wait_asynccnt 0x0" ::: "memory");
}

// ---------------------------------------------------------------------------
// Tiled WMMA GEMM: D[M,N] = epilogue( A[M,K] @ Bm[N,K]^T )
// 128 threads = 4 waves. Block tile 64 x (64*NW); wave tile 64 x (16*NW).
// K staged via double-buffered LDS in slabs of 64 (two WMMA K-steps/slab),
// filled with global_load_async_to_lds_b128 overlapped with WMMA compute.
// ---------------------------------------------------------------------------
template<int EPI, int NW>
__global__ __launch_bounds__(128)
void gemm_wmma_kernel(GP p) {
  const int z    = blockIdx.z;
  const int m0   = blockIdx.y * 64;
  const int n0   = blockIdx.x * (64 * NW);
  const int tid  = threadIdx.x;
  const int lane = tid & 31;
  const int w    = tid >> 5;

  const _Float16* A  = p.A  + (long long)(z / p.zdiv) * p.sAhi
                            + (long long)(z % p.zdiv) * p.sAlo;
  const _Float16* Bm = p.Bm + (long long)z * p.sBz;

  __shared__ __attribute__((aligned(16))) _Float16 As[2][64 * LDS_STRIDE];
  __shared__ __attribute__((aligned(16))) _Float16 Bs[2][64 * NW * LDS_STRIDE];

  v8f acc[4][NW];
  #pragma unroll
  for (int i = 0; i < 4; i++)
    #pragma unroll
    for (int jn = 0; jn < NW; jn++)
      #pragma unroll
      for (int e = 0; e < 8; e++) acc[i][jn][e] = 0.0f;

  union Frag { v16h v; uint4 q[2]; };

  // Issue all async B128 copies for one 64-wide K slab into LDS buffer `buf`.
  auto issue_slab = [&](int kt, int buf) {
    #pragma unroll
    for (int c = 0; c < 4; c++) {
      const int g = tid * 4 + c;
      const int r = g >> 3, col = (g & 7) * 8;
      const unsigned lds =
          (unsigned)(unsigned long long)(const void*)&As[buf][r * LDS_STRIDE + col];
      async_load_b128(lds, (const void*)(A + (long long)(m0 + r) * p.lda + kt + col));
    }
    #pragma unroll
    for (int c = 0; c < 4 * NW; c++) {
      const int g = tid * 4 * NW + c;
      const int r = g >> 3, col = (g & 7) * 8;
      const unsigned lds =
          (unsigned)(unsigned long long)(const void*)&Bs[buf][r * LDS_STRIDE + col];
      async_load_b128(lds, (const void*)(Bm + (long long)(n0 + r) * p.ldb + kt + col));
    }
  };

  // Prologue: fill buffer 0.
  issue_slab(0, 0);
  wait_async0();
  __syncthreads();

  int cur = 0;
  for (int kt = 0; kt < p.K; kt += 64) {
    // Start fetching the next slab into the alternate buffer (overlaps WMMA).
    if (kt + 64 < p.K) issue_slab(kt + 64, cur ^ 1);

    // ---- two WMMA K-steps on the current slab ----
    #pragma unroll
    for (int k2 = 0; k2 < 64; k2 += 32) {
      // B fragments (32x16 f16): lane = column, 16 contiguous K halfs at
      // k2 + 16*(lane/16).
      Frag bf[NW];
      #pragma unroll
      for (int jn = 0; jn < NW; jn++) {
        const int bn = w * 16 * NW + jn * 16 + (lane & 15);
        const int bc = k2 + ((lane >> 4) << 4);
        bf[jn].q[0] = *(const uint4*)&Bs[cur][bn * LDS_STRIDE + bc];
        bf[jn].q[1] = *(const uint4*)&Bs[cur][bn * LDS_STRIDE + bc + 8];
      }
      // A fragments (16x32 f16): 16 contiguous K halfs
      // {base..base+15}, base = k2 + 16*(lane/16).
      const int abase = k2 + ((lane >> 4) << 4);
      #pragma unroll
      for (int i = 0; i < 4; i++) {
        Frag af;
        const int ar = (i << 4) + (lane & 15);
        af.q[0] = *(const uint4*)&As[cur][ar * LDS_STRIDE + abase];
        af.q[1] = *(const uint4*)&As[cur][ar * LDS_STRIDE + abase + 8];
        #pragma unroll
        for (int jn = 0; jn < NW; jn++)
          acc[i][jn] = __builtin_amdgcn_wmma_f32_16x16x32_f16(
              false, af.v, false, bf[jn].v, (short)0, acc[i][jn], false, false);
      }
    }

    // Next slab must have landed (own-wave ASYNCcnt) and all waves must be
    // done reading `cur` (barrier) before it can be overwritten next iter.
    wait_async0();
    __syncthreads();
    cur ^= 1;
  }

  // ---- epilogue ----
  // element e of acc[i][jn]: row = m0+16*i+8*(lane/16)+e,
  //                          col = n0+w*16*NW+jn*16+lane%16
  #pragma unroll
  for (int i = 0; i < 4; i++) {
    #pragma unroll
    for (int jn = 0; jn < NW; jn++) {
      #pragma unroll
      for (int e = 0; e < 8; e++) {
        float val = acc[i][jn][e];
        const int mg = m0 + (i << 4) + ((lane >> 4) << 3) + e;
        const int ng = n0 + w * 16 * NW + jn * 16 + (lane & 15);
        if constexpr (EPI == EPI_PLAIN) {
          ((_Float16*)p.D)[(long long)mg * p.ldd + ng] = (_Float16)val;
        } else if constexpr (EPI == EPI_VSPLIT) {
          if (p.mask && !p.mask[mg]) val = 0.0f;
          const int bb = mg >> 12, nn = mg & (NSEQ - 1);
          const int hh = ng >> 6,  dd = ng & (DHEAD - 1);
          ((_Float16*)p.D)[(((long long)(bb * HEADS + hh)) * DHEAD + dd) * NSEQ + nn] =
              (_Float16)val;
        } else if constexpr (EPI == EPI_FEAT) {
          val = fmaxf(val, 0.0f) + EPS_K;
          if (ng >= MFEAT) val = 0.0f;
          ((_Float16*)p.D)[(long long)z * p.sDz + (long long)mg * MPAD + ng] = (_Float16)val;
        } else if constexpr (EPI == EPI_FEATT) {
          val = fmaxf(val, 0.0f) + EPS_K;
          if (ng >= MFEAT) val = 0.0f;
          ((_Float16*)p.D)[(long long)z * p.sDz + (long long)ng * NSEQ + mg] = (_Float16)val;
        } else if constexpr (EPI == EPI_CTXT) {
          ((_Float16*)p.D)[(long long)z * p.sDz + (long long)ng * MPAD + mg] = (_Float16)val;
        } else if constexpr (EPI == EPI_OUTSCALE) {
          val *= p.dinv[(long long)z * NSEQ + mg];
          const int bb = z >> 4, hh = z & 15;
          ((_Float16*)p.D)[((long long)(bb * NSEQ + mg)) * CDIM + hh * DHEAD + ng] =
              (_Float16)val;
        } else if constexpr (EPI == EPI_BIASRES) {
          val += p.bias[ng] + p.res[(long long)mg * p.ldd + ng];
          ((float*)p.D)[(long long)mg * p.ldd + ng] = val;
        } else if constexpr (EPI == EPI_GELU) {
          const float t = val + p.bias[ng];
          val = 0.5f * t * (1.0f + erff(t * 0.70710678118654752f));
          ((_Float16*)p.D)[(long long)mg * p.ldd + ng] = (_Float16)val;
        }
      }
    }
  }
}

// ---------------------------------------------------------------------------
// LayerNorm over C=1024, one block per row, f16 output.
// ---------------------------------------------------------------------------
__global__ __launch_bounds__(256)
void ln_f16_kernel(const float* __restrict__ x, const float* __restrict__ g,
                   const float* __restrict__ b, _Float16* __restrict__ out) {
  const long long row = blockIdx.x;
  const float4* xr = (const float4*)(x + row * CDIM);
  const int tid = threadIdx.x;
  float4 v = xr[tid];
  float s  = v.x + v.y + v.z + v.w;
  float ss = v.x * v.x + v.y * v.y + v.z * v.z + v.w * v.w;
  for (int o = 16; o; o >>= 1) { s += __shfl_xor(s, o); ss += __shfl_xor(ss, o); }
  __shared__ float sw[8], ssw[8];
  const int w = tid >> 5, lane = tid & 31;
  if (lane == 0) { sw[w] = s; ssw[w] = ss; }
  __syncthreads();
  float ts = 0.f, tss = 0.f;
  for (int i = 0; i < 8; i++) { ts += sw[i]; tss += ssw[i]; }
  const float mu  = ts * (1.0f / CDIM);
  const float var = tss * (1.0f / CDIM) - mu * mu;
  const float rs  = rsqrtf(var + LN_EPS);
  const int c = tid * 4;
  _Float16* o4 = out + row * CDIM + c;
  o4[0] = (_Float16)((v.x - mu) * rs * g[c + 0] + b[c + 0]);
  o4[1] = (_Float16)((v.y - mu) * rs * g[c + 1] + b[c + 1]);
  o4[2] = (_Float16)((v.z - mu) * rs * g[c + 2] + b[c + 2]);
  o4[3] = (_Float16)((v.w - mu) * rs * g[c + 3] + b[c + 3]);
}

// kfsum[row] = sum over rowlen of f16 row (kfT rows are n-contiguous)
__global__ __launch_bounds__(256)
void rowsum_kernel(const _Float16* __restrict__ src, float* __restrict__ dst,
                   int rowlen) {
  const long long row = blockIdx.x;
  const _Float16* p = src + row * rowlen;
  const int tid = threadIdx.x;
  float s = 0.f;
  for (int i = tid; i < rowlen; i += 256) s += (float)p[i];
  for (int o = 16; o; o >>= 1) s += __shfl_xor(s, o);
  __shared__ float sw[8];
  if ((tid & 31) == 0) sw[tid >> 5] = s;
  __syncthreads();
  if (tid == 0) { float t = 0.f; for (int i = 0; i < 8; i++) t += sw[i]; dst[row] = t; }
}

// dinv[b,h,n] = 1 / dot(qf[b,h,n,:], kfsum[b,h,:]) ; one wave per row
__global__ __launch_bounds__(256)
void dinv_kernel(const _Float16* __restrict__ qf, const float* __restrict__ kfsum,
                 float* __restrict__ dinv) {
  const int w = threadIdx.x >> 5, lane = threadIdx.x & 31;
  const long long r = (long long)blockIdx.x * 8 + w;
  const _Float16* q = qf + r * MPAD;
  const float* ks = kfsum + (r >> 12) * MPAD;   // r>>12: (b*H+h) index
  float s = 0.f;
  for (int i = lane; i < MPAD; i += 32) s += (float)q[i] * ks[i];
  for (int o = 16; o; o >>= 1) s += __shfl_xor(s, o);
  if (lane == 0) dinv[r] = 1.0f / s;
}

__global__ void cvt_f16_kernel(const float* __restrict__ s, _Float16* __restrict__ d,
                               long long n) {
  for (long long i = (long long)blockIdx.x * blockDim.x + threadIdx.x; i < n;
       i += (long long)gridDim.x * blockDim.x)
    d[i] = (_Float16)s[i];
}

__global__ void zero_f16_kernel(_Float16* __restrict__ d, long long n) {
  for (long long i = (long long)blockIdx.x * blockDim.x + threadIdx.x; i < n;
       i += (long long)gridDim.x * blockDim.x)
    d[i] = (_Float16)0.0f;
}

// ---------------------------------------------------------------------------
extern "C" void kernel_launch(void* const* d_in, const int* in_sizes, int n_in,
                              void* d_out, int out_size, void* d_ws, size_t ws_size,
                              hipStream_t stream) {
  const float* x    = (const float*)d_in[0];
  const unsigned char* mask = (const unsigned char*)d_in[1];
  const float* g1   = (const float*)d_in[2];
  const float* b1   = (const float*)d_in[3];
  const float* g2   = (const float*)d_in[4];
  const float* b2   = (const float*)d_in[5];
  const float* Wq   = (const float*)d_in[6];
  const float* Wk   = (const float*)d_in[7];
  const float* Wv   = (const float*)d_in[8];
  const float* Wo   = (const float*)d_in[9];
  const float* bo   = (const float*)d_in[10];
  const float* proj = (const float*)d_in[11];
  const float* W1   = (const float*)d_in[12];
  const float* b1m  = (const float*)d_in[13];
  const float* W2   = (const float*)d_in[14];
  const float* b2m  = (const float*)d_in[15];
  float* out = (float*)d_out;

  char* ws = (char*)d_ws;
  size_t off = 0;
  auto alloc = [&](size_t bytes) -> char* {
    char* r = ws + off;
    off += (bytes + 255) & ~(size_t)255;
    return r;
  };
  _Float16* Wq16  = (_Float16*)alloc((size_t)CDIM * CDIM * 2);
  _Float16* Wk16  = (_Float16*)alloc((size_t)CDIM * CDIM * 2);
  _Float16* Wv16  = (_Float16*)alloc((size_t)CDIM * CDIM * 2);
  _Float16* Wo16  = (_Float16*)alloc((size_t)CDIM * CDIM * 2);
  _Float16* W116  = (_Float16*)alloc((size_t)4 * CDIM * CDIM * 2);
  _Float16* W216  = (_Float16*)alloc((size_t)4 * CDIM * CDIM * 2);
  _Float16* projh = (_Float16*)alloc((size_t)MPAD * DHEAD * 2);
  _Float16* xh    = (_Float16*)alloc((size_t)BN * CDIM * 2);
  _Float16* qh    = (_Float16*)alloc((size_t)BN * CDIM * 2);
  _Float16* kh    = (_Float16*)alloc((size_t)BN * CDIM * 2);
  _Float16* vT    = (_Float16*)alloc((size_t)BN * CDIM * 2);
  _Float16* qf    = (_Float16*)alloc((size_t)BDIM * HEADS * NSEQ * MPAD * 2);
  _Float16* kfT   = (_Float16*)alloc((size_t)BDIM * HEADS * NSEQ * MPAD * 2);
  _Float16* ctxT  = (_Float16*)alloc((size_t)BDIM * HEADS * DHEAD * MPAD * 2);
  float*    kfsum = (float*)alloc((size_t)BDIM * HEADS * MPAD * 4);
  float*    dinv  = (float*)alloc((size_t)BDIM * HEADS * NSEQ * 4);
  // Aliased (producers strictly after last consumer of the aliased buffer):
  _Float16* o16 = kh;            // attention output, f16 [BN, C]
  float*    x2  = (float*)kfT;   // post-attn residual, f32 [BN, C]
  _Float16* x2h = qh;            // LN2 output, f16 [BN, C]
  _Float16* h1  = qf;            // MLP hidden, f16 [BN, 4C]

  // 0) weight conversion + padded projection
  cvt_f16_kernel<<<dim3(2048), 256, 0, stream>>>(Wq, Wq16, (long long)CDIM * CDIM);
  cvt_f16_kernel<<<dim3(2048), 256, 0, stream>>>(Wk, Wk16, (long long)CDIM * CDIM);
  cvt_f16_kernel<<<dim3(2048), 256, 0, stream>>>(Wv, Wv16, (long long)CDIM * CDIM);
  cvt_f16_kernel<<<dim3(2048), 256, 0, stream>>>(Wo, Wo16, (long long)CDIM * CDIM);
  cvt_f16_kernel<<<dim3(4096), 256, 0, stream>>>(W1, W116, (long long)4 * CDIM * CDIM);
  cvt_f16_kernel<<<dim3(4096), 256, 0, stream>>>(W2, W216, (long long)4 * CDIM * CDIM);
  zero_f16_kernel<<<dim3(80), 256, 0, stream>>>(projh, (long long)MPAD * DHEAD);
  cvt_f16_kernel<<<dim3(80), 256, 0, stream>>>(proj, projh, (long long)MFEAT * DHEAD);

  // 1) LN1
  ln_f16_kernel<<<dim3(BN), 256, 0, stream>>>(x, g1, b1, xh);

  GP p{};
  p.zdiv = 1;

  // 2) Q, K projections (f16 plain)   [M=16384, N=1024, K=1024]
  p.A = xh; p.lda = CDIM; p.ldb = CDIM; p.ldd = CDIM;
  p.M = BN; p.N = CDIM; p.K = CDIM;
  p.Bm = Wq16; p.D = qh;
  gemm_wmma_kernel<EPI_PLAIN, 2><<<dim3(8, 256, 1), 128, 0, stream>>>(p);
  p.Bm = Wk16; p.D = kh;
  gemm_wmma_kernel<EPI_PLAIN, 2><<<dim3(8, 256, 1), 128, 0, stream>>>(p);
  // 3) V projection, masked, scattered to vT[b,h,d,n]
  p.Bm = Wv16; p.D = vT; p.mask = mask;
  gemm_wmma_kernel<EPI_VSPLIT, 2><<<dim3(8, 256, 1), 128, 0, stream>>>(p);
  p.mask = nullptr;

  // 4) qf = relu(q @ projh^T)+eps  [per (b,h) batch, M=4096, N=320, K=64]
  p.A = qh; p.lda = CDIM; p.sAhi = (long long)NSEQ * CDIM; p.sAlo = DHEAD; p.zdiv = HEADS;
  p.Bm = projh; p.ldb = DHEAD; p.sBz = 0;
  p.M = NSEQ; p.N = MPAD; p.K = DHEAD;
  p.D = qf; p.sDz = (long long)NSEQ * MPAD;
  gemm_wmma_kernel<EPI_FEAT, 1><<<dim3(5, 64, 64), 128, 0, stream>>>(p);
  // 5) kfT (transposed store)
  p.A = kh; p.D = kfT;
  gemm_wmma_kernel<EPI_FEATT, 1><<<dim3(5, 64, 64), 128, 0, stream>>>(p);

  // 6) kfsum[b,h,m] = sum_n kfT
  rowsum_kernel<<<dim3(BDIM * HEADS * MPAD), 256, 0, stream>>>(kfT, kfsum, NSEQ);

  // 7) ctxT[b,h,d,m] = (kfT @ vT^T)^T  [M=320, N=64, K=4096 per head]
  p.A = kfT; p.lda = NSEQ; p.sAhi = (long long)MPAD * NSEQ; p.sAlo = 0; p.zdiv = 1;
  p.Bm = vT; p.ldb = NSEQ; p.sBz = (long long)DHEAD * NSEQ;
  p.M = MPAD; p.N = DHEAD; p.K = NSEQ;
  p.D = ctxT; p.sDz = (long long)DHEAD * MPAD;
  gemm_wmma_kernel<EPI_CTXT, 1><<<dim3(1, 5, 64), 128, 0, stream>>>(p);

  // 8) dinv
  dinv_kernel<<<dim3(BDIM * HEADS * NSEQ / 8), 256, 0, stream>>>(qf, kfsum, dinv);

  // 9) o = (qf @ ctxT^T) * dinv, merged heads  [M=4096, N=64, K=320 per head]
  p.A = qf; p.lda = MPAD; p.sAhi = (long long)NSEQ * MPAD; p.sAlo = 0;
  p.Bm = ctxT; p.ldb = MPAD; p.sBz = (long long)DHEAD * MPAD;
  p.M = NSEQ; p.N = DHEAD; p.K = MPAD;
  p.D = o16; p.dinv = dinv;
  gemm_wmma_kernel<EPI_OUTSCALE, 1><<<dim3(1, 64, 64), 128, 0, stream>>>(p);
  p.dinv = nullptr;

  // 10) x2 = x + o @ Wo^T + bo  (f32)  [M=16384, N=1024, K=1024]
  p.A = o16; p.lda = CDIM; p.sAhi = 0; p.sBz = 0; p.sDz = 0;
  p.Bm = Wo16; p.ldb = CDIM; p.ldd = CDIM;
  p.M = BN; p.N = CDIM; p.K = CDIM;
  p.D = x2; p.bias = bo; p.res = x;
  gemm_wmma_kernel<EPI_BIASRES, 2><<<dim3(8, 256, 1), 128, 0, stream>>>(p);

  // 11) LN2
  ln_f16_kernel<<<dim3(BN), 256, 0, stream>>>(x2, g2, b2, x2h);

  // 12) h1 = gelu(x2h @ W1^T + b1m)  [M=16384, N=4096, K=1024]
  p.A = x2h; p.lda = CDIM;
  p.Bm = W116; p.ldb = CDIM; p.ldd = 4 * CDIM;
  p.M = BN; p.N = 4 * CDIM; p.K = CDIM;
  p.D = h1; p.bias = b1m; p.res = nullptr;
  gemm_wmma_kernel<EPI_GELU, 2><<<dim3(32, 256, 1), 128, 0, stream>>>(p);

  // 13) out = x2 + h1 @ W2^T + b2m  (f32)  [M=16384, N=1024, K=4096]
  p.A = h1; p.lda = 4 * CDIM;
  p.Bm = W216; p.ldb = 4 * CDIM; p.ldd = CDIM;
  p.M = BN; p.N = CDIM; p.K = 4 * CDIM;
  p.D = out; p.bias = b2m; p.res = x2;
  gemm_wmma_kernel<EPI_BIASRES, 2><<<dim3(8, 256, 1), 128, 0, stream>>>(p);
}
